// ContextGuidedDeformableAttention_81406810129135
// MI455X (gfx1250) — matrix operand order repaired and checked by reading, not compiled, not run on hardware
//
#include <hip/hip_runtime.h>
#include <math.h>

// ContextGuidedDeformableAttention for MI455X (gfx1250, wave32).
// Heavy GEMMs (W_q, W_pre, W_k, W_v, W_p) -> v_wmma_f32_16x16x32_f16 with
// fragment-native LDS layout (80B row pitch => conflict-free ds_load_b128)
// and double-buffered software pipeline.
// Gather/softmax/LN/pool stages -> VALU kernels with layouts chosen so the
// deformable sampling wave (one wave per (b,head,pixel), lane = head-dim)
// issues coalesced 128B reads.

#define BB 8
#define HH 48
#define WW 48
#define NPIX 2304        // H*W
#define C384 384
#define NHEAD 12
#define NPTS 4
#define HD 32
#define SCALE 0.17677669529663687f   // 32^-0.5

typedef __attribute__((ext_vector_type(16))) _Float16 v16h;
typedef __attribute__((ext_vector_type(8)))  _Float16 v8h;
typedef __attribute__((ext_vector_type(8)))  float    v8f;

__device__ __forceinline__ float gelu_f(float x) {
    return 0.5f * x * (1.0f + erff(x * 0.7071067811865476f));
}

// ---------------------------------------------------------------------------
// WMMA GEMM:  Out[b,o,n] = epi( sum_c W[o,c] * X[b,c,n] )
//   IN_NC : 0 -> X is (B, Cin, NPIX);  1 -> X is (B, NPIX, Cin)
//   OUT_NC: 0 -> Out is (B, O, NPIX);  1 -> Out is (B, NPIX, O)
//   EPI   : 0 none, 1 GELU, 2 BatchNorm (e0=g, e1=b, e2=mean, e3=var)
// Block = 256 threads (8 waves); tile = 64 (o) x 128 (n); K staged 32-deep,
// LDS double-buffered, rows padded to 40 halves (80B) so every fragment is
// two aligned, bank-conflict-free ds_load_b128.
// ---------------------------------------------------------------------------
#define TILE_O 64
#define TILE_N 128
#define KC 32
#define KPAD 40

template<int IN_NC, int OUT_NC, int EPI>
__global__ __launch_bounds__(256) void gemm_wmma(
    const float* __restrict__ X, const float* __restrict__ Wt,
    float* __restrict__ Out, int Cin, int Ochan,
    const float* __restrict__ e0, const float* __restrict__ e1,
    const float* __restrict__ e2, const float* __restrict__ e3)
{
    __shared__ alignas(16) _Float16 lds_w[2][TILE_O][KPAD];
    __shared__ alignas(16) _Float16 lds_x[2][TILE_N][KPAD];

    const int b    = blockIdx.z;
    const int o0   = blockIdx.y * TILE_O;
    const int n0   = blockIdx.x * TILE_N;
    const int t    = threadIdx.x;
    const int lane = t & 31;
    const int wv   = t >> 5;

    // per-thread staging assignments
    const int w_ol = t >> 2;          // 0..63
    const int w_cl = (t & 3) * 8;     // 0,8,16,24
    const int x_nl = t >> 1;          // 0..127
    const int x_cl = (t & 1) * 16;    // 0 or 16

    float wreg[8];
    float xreg[16];

    auto load_tile = [&](int kc) {
        const float* wsrc = Wt + (size_t)(o0 + w_ol) * Cin + kc + w_cl;
#pragma unroll
        for (int i = 0; i < 8; ++i) wreg[i] = wsrc[i];
        if (IN_NC == 0) {
            const float* xsrc = X + ((size_t)b * Cin + kc + x_cl) * NPIX + n0 + x_nl;
#pragma unroll
            for (int i = 0; i < 16; ++i) xreg[i] = xsrc[(size_t)i * NPIX];
        } else {
            const float* xsrc = X + ((size_t)b * NPIX + n0 + x_nl) * Cin + kc + x_cl;
#pragma unroll
            for (int i = 0; i < 16; ++i) xreg[i] = xsrc[i];
        }
    };
    auto store_tile = [&](int buf) {
        v8h wh;
#pragma unroll
        for (int i = 0; i < 8; ++i) wh[i] = (_Float16)wreg[i];
        *(v8h*)&lds_w[buf][w_ol][w_cl] = wh;
        v8h x0, x1;
#pragma unroll
        for (int i = 0; i < 8; ++i) { x0[i] = (_Float16)xreg[i]; x1[i] = (_Float16)xreg[8 + i]; }
        *(v8h*)&lds_x[buf][x_nl][x_cl]     = x0;
        *(v8h*)&lds_x[buf][x_nl][x_cl + 8] = x1;
    };

    v8f acc[4];
#pragma unroll
    for (int j = 0; j < 4; ++j)
#pragma unroll
        for (int e = 0; e < 8; ++e) acc[j][e] = 0.0f;

    load_tile(0);
    store_tile(0);
    __syncthreads();

    const int nsteps = Cin / KC;
    const int orow = (wv & 3) * 16 + (lane & 15);
    const int kb   = (lane < 16) ? 0 : 8;    // A: K halves {kb..kb+7, 16+kb..}
    const int kb2  = (lane < 16) ? 0 : 16;   // B: K halves {kb..kb+7, 16+kb..} (same as A)

    for (int s = 0; s < nsteps; ++s) {
        const int cur = s & 1;
        const bool has_next = (s + 1 < nsteps);
        if (has_next) load_tile((s + 1) * KC);

        // A fragment: two aligned 16B LDS reads
        v8h alo = *(const v8h*)&lds_w[cur][orow][kb];
        v8h ahi = *(const v8h*)&lds_w[cur][orow][16 + kb];
        v16h a = __builtin_shufflevector(alo, ahi,
                    0,1,2,3,4,5,6,7,8,9,10,11,12,13,14,15);
#pragma unroll
        for (int j = 0; j < 4; ++j) {
            int ncol = ((wv >> 2) << 6) + (j << 4) + (lane & 15);
            v8h blo = *(const v8h*)&lds_x[cur][ncol][kb];
            v8h bhi = *(const v8h*)&lds_x[cur][ncol][16 + kb];
            v16h bf = __builtin_shufflevector(blo, bhi,
                        0,1,2,3,4,5,6,7,8,9,10,11,12,13,14,15);
            acc[j] = __builtin_amdgcn_wmma_f32_16x16x32_f16(
                false, a, false, bf, (short)0, acc[j], false, false);
        }

        if (has_next) store_tile(cur ^ 1);
        __syncthreads();
    }

    // Epilogue. D layout: VGPR r, M = r + 8*(lane>=16), N = lane%16.
#pragma unroll
    for (int j = 0; j < 4; ++j) {
        int n = n0 + ((wv >> 2) << 6) + (j << 4) + (lane & 15);
#pragma unroll
        for (int r = 0; r < 8; ++r) {
            int o = o0 + (wv & 3) * 16 + r + ((lane >> 4) << 3);
            float y = acc[j][r];
            if (EPI == 1) {
                y = gelu_f(y);
            } else if (EPI == 2) {
                float sc = e0[o] * rsqrtf(e3[o] + 1e-5f);
                y = y * sc + (e1[o] - e2[o] * sc);
            }
            if (OUT_NC == 0) Out[((size_t)b * Ochan + o) * NPIX + n] = y;
            else             Out[((size_t)b * NPIX  + n) * Ochan + o] = y;
        }
    }
}

// ---------------------------------------------------------------------------
// 7x7 adaptive average pool of cg (B,C,H,W) -> pooled (B,C,7,7)
// ---------------------------------------------------------------------------
__global__ void pool_kernel(const float* __restrict__ cg, float* __restrict__ pooled)
{
    int bc = blockIdx.x;           // 0 .. B*384-1
    int t  = threadIdx.x;
    if (t >= 49) return;
    int i = t / 7, j = t % 7;
    int hs = i * HH / 7, he = ((i + 1) * HH + 6) / 7;
    int ws = j * WW / 7, we = ((j + 1) * WW + 6) / 7;
    const float* src = cg + (size_t)bc * NPIX;
    float s = 0.0f;
    for (int h = hs; h < he; ++h)
        for (int w = ws; w < we; ++w) s += src[h * WW + w];
    pooled[(size_t)bc * 49 + t] = s / (float)((he - hs) * (we - ws));
}

// ---------------------------------------------------------------------------
// LayerNorm over channels (eps 1e-6) in-place on pooled (B,384,49)
// ---------------------------------------------------------------------------
__global__ __launch_bounds__(128) void ln1_kernel(
    float* __restrict__ pooled, const float* __restrict__ g, const float* __restrict__ bt)
{
    int b = blockIdx.x / 49, ij = blockIdx.x % 49;
    int t = threadIdx.x;
    __shared__ float red[128];
    float v[3];
    float s = 0.0f;
#pragma unroll
    for (int i = 0; i < 3; ++i) {
        v[i] = pooled[((size_t)(b * C384 + t + i * 128)) * 49 + ij];
        s += v[i];
    }
    red[t] = s; __syncthreads();
    for (int st = 64; st > 0; st >>= 1) { if (t < st) red[t] += red[t + st]; __syncthreads(); }
    float mu = red[0] * (1.0f / 384.0f);
    __syncthreads();
    float s2 = 0.0f;
#pragma unroll
    for (int i = 0; i < 3; ++i) { float d = v[i] - mu; s2 += d * d; }
    red[t] = s2; __syncthreads();
    for (int st = 64; st > 0; st >>= 1) { if (t < st) red[t] += red[t + st]; __syncthreads(); }
    float inv = rsqrtf(red[0] * (1.0f / 384.0f) + 1e-6f);
#pragma unroll
    for (int i = 0; i < 3; ++i) {
        int c = t + i * 128;
        pooled[((size_t)(b * C384 + c)) * 49 + ij] = (v[i] - mu) * inv * g[c] + bt[c];
    }
}

// W_post: guide_small[b,o,ij] = sum_c Wpost[o,c] * lnp[b,c,ij]   (32x384, tiny)
__global__ void post_kernel(const float* __restrict__ lnp, const float* __restrict__ Wp,
                            float* __restrict__ gs)
{
    int idx = blockIdx.x * blockDim.x + threadIdx.x;
    if (idx >= BB * 32 * 49) return;
    int ij = idx % 49, o = (idx / 49) % 32, b = idx / (49 * 32);
    float s = 0.0f;
    for (int c = 0; c < C384; ++c)
        s += Wp[o * C384 + c] * lnp[((size_t)(b * C384 + c)) * 49 + ij];
    gs[idx] = s;
}

// Bilinear 7 -> 48 resize of guide_small -> guide (B,32,NPIX)
__global__ void resize_kernel(const float* __restrict__ gs, float* __restrict__ guide)
{
    int idx = blockIdx.x * blockDim.x + threadIdx.x;
    if (idx >= BB * 32 * NPIX) return;
    int n = idx % NPIX;
    int bo = idx / NPIX;
    int h = n / WW, w = n % WW;
    float sy = (h + 0.5f) * (7.0f / 48.0f) - 0.5f;
    sy = fminf(fmaxf(sy, 0.0f), 6.0f);
    float sx = (w + 0.5f) * (7.0f / 48.0f) - 0.5f;
    sx = fminf(fmaxf(sx, 0.0f), 6.0f);
    int y0 = (int)floorf(sy); int y1 = y0 + 1 > 6 ? 6 : y0 + 1; float fy = sy - (float)y0;
    int x0 = (int)floorf(sx); int x1 = x0 + 1 > 6 ? 6 : x0 + 1; float fx = sx - (float)x0;
    const float* base = gs + (size_t)bo * 49;
    float v = (1.0f - fy) * ((1.0f - fx) * base[y0 * 7 + x0] + fx * base[y0 * 7 + x1])
            +         fy  * ((1.0f - fx) * base[y1 * 7 + x0] + fx * base[y1 * 7 + x1]);
    guide[idx] = v;
}

// Depthwise 3x3 (correlation, pad 1) + bias -> dw (B,384,NPIX)
__global__ void dwconv_kernel(const float* __restrict__ x, const float* __restrict__ wt,
                              const float* __restrict__ bias, float* __restrict__ outp)
{
    int idx = blockIdx.x * blockDim.x + threadIdx.x;
    if (idx >= BB * C384 * NPIX) return;
    int n = idx % NPIX;
    int c = (idx / NPIX) % C384;
    int b = idx / (NPIX * C384);
    int h = n / WW, w = n % WW;
    const float* xp = x + ((size_t)(b * C384) + c) * NPIX;
    const float* wp = wt + c * 9;
    float s = bias[c];
#pragma unroll
    for (int dy = -1; dy <= 1; ++dy) {
        int hh = h + dy; if (hh < 0 || hh >= HH) continue;
#pragma unroll
        for (int dx = -1; dx <= 1; ++dx) {
            int ww2 = w + dx; if (ww2 < 0 || ww2 >= WW) continue;
            s += wp[(dy + 1) * 3 + (dx + 1)] * xp[hh * WW + ww2];
        }
    }
    outp[idx] = s;
}

// LayerNorm over channels (eps 1e-6) + GELU on dw -> lo_pre (B,384,NPIX)
__global__ __launch_bounds__(128) void ln2gelu_kernel(
    const float* __restrict__ dw, const float* __restrict__ g,
    const float* __restrict__ bt, float* __restrict__ outp)
{
    int b = blockIdx.x / NPIX, n = blockIdx.x % NPIX;
    int t = threadIdx.x;
    __shared__ float red[128];
    float v[3];
    float s = 0.0f;
#pragma unroll
    for (int i = 0; i < 3; ++i) {
        v[i] = dw[((size_t)(b * C384) + (t + i * 128)) * NPIX + n];
        s += v[i];
    }
    red[t] = s; __syncthreads();
    for (int st = 64; st > 0; st >>= 1) { if (t < st) red[t] += red[t + st]; __syncthreads(); }
    float mu = red[0] * (1.0f / 384.0f);
    __syncthreads();
    float s2 = 0.0f;
#pragma unroll
    for (int i = 0; i < 3; ++i) { float d = v[i] - mu; s2 += d * d; }
    red[t] = s2; __syncthreads();
    for (int st = 64; st > 0; st >>= 1) { if (t < st) red[t] += red[t + st]; __syncthreads(); }
    float inv = rsqrtf(red[0] * (1.0f / 384.0f) + 1e-6f);
#pragma unroll
    for (int i = 0; i < 3; ++i) {
        int c = t + i * 128;
        float xn = (v[i] - mu) * inv * g[c] + bt[c];
        outp[((size_t)(b * C384) + c) * NPIX + n] = gelu_f(xn);
    }
}

// W_lo (32x384) + b_lo -> lo (B,32,NPIX)
__global__ void lo_kernel(const float* __restrict__ lp, const float* __restrict__ Wlo,
                          const float* __restrict__ blo, float* __restrict__ lo)
{
    int idx = blockIdx.x * blockDim.x + threadIdx.x;
    if (idx >= BB * 32 * NPIX) return;
    int n = idx % NPIX, o = (idx / NPIX) % 32, b = idx / (NPIX * 32);
    float s = blo[o];
    const float* xp = lp + (size_t)(b * C384) * NPIX + n;
    const float* wp = Wlo + o * C384;
    for (int c = 0; c < C384; ++c) s += wp[c] * xp[(size_t)c * NPIX];
    lo[idx] = s;
}

// W_off (96x64) on concat(guide, lo) + b_off -> off (B,96,NPIX)
__global__ void off_kernel(const float* __restrict__ guide, const float* __restrict__ lo,
                           const float* __restrict__ Woff, const float* __restrict__ boff,
                           float* __restrict__ off)
{
    int idx = blockIdx.x * blockDim.x + threadIdx.x;
    if (idx >= BB * 96 * NPIX) return;
    int n = idx % NPIX, o = (idx / NPIX) % 96, b = idx / (NPIX * 96);
    float s = boff[o];
    const float* wp = Woff + o * 64;
    const float* gp = guide + (size_t)(b * 32) * NPIX + n;
    const float* lp = lo    + (size_t)(b * 32) * NPIX + n;
    for (int c = 0; c < 32; ++c) s += wp[c]      * gp[(size_t)c * NPIX];
    for (int c = 0; c < 32; ++c) s += wp[32 + c] * lp[(size_t)c * NPIX];
    off[idx] = s;
}

// ---------------------------------------------------------------------------
// Deformable sampling + biased softmax attention.
// One wave per (b, head, n); lane = head-dim d. q/k/v in (B,N,384) layout so
// every corner fetch is 32 consecutive floats (one 128B line per corner).
// ---------------------------------------------------------------------------
__global__ __launch_bounds__(256) void attn_kernel(
    const float* __restrict__ qT, const float* __restrict__ kT,
    const float* __restrict__ vT, const float* __restrict__ off,
    const float* __restrict__ ab, float* __restrict__ out)
{
    int gid  = blockIdx.x * blockDim.x + threadIdx.x;
    int wave = gid >> 5;
    int lane = gid & 31;
    if (wave >= BB * NHEAD * NPIX) return;
    int n    = wave % NPIX;
    int head = (wave / NPIX) % NHEAD;
    int b    = wave / (NPIX * NHEAD);
    int qx = n % WW, qy = n / WW;

    float qd = qT[((size_t)(b * NPIX + n)) * C384 + head * HD + lane];

    float sc[NPTS], vs[NPTS];
#pragma unroll
    for (int p = 0; p < NPTS; ++p) {
        int co = head * 8 + p * 2;
        float ox = off[((size_t)(b * 96 + co))     * NPIX + n];
        float oy = off[((size_t)(b * 96 + co + 1)) * NPIX + n];
        float gx = ((qx + 0.5f) / 48.0f + ox / 48.0f) * 2.0f - 1.0f;
        float gy = ((qy + 0.5f) / 48.0f + oy / 48.0f) * 2.0f - 1.0f;
        float x = (gx + 1.0f) * 0.5f * 48.0f - 0.5f;
        float y = (gy + 1.0f) * 0.5f * 48.0f - 0.5f;
        float x0f = floorf(x), y0f = floorf(y);
        int x0 = (int)x0f, y0 = (int)y0f;
        float fx = x - x0f, fy = y - y0f;
        float kacc = 0.0f, vacc = 0.0f;
#pragma unroll
        for (int cy = 0; cy < 2; ++cy)
#pragma unroll
            for (int cx = 0; cx < 2; ++cx) {
                int ix = x0 + cx, iy = y0 + cy;
                if (ix >= 0 && ix < WW && iy >= 0 && iy < HH) {
                    float wgt = (cx ? fx : 1.0f - fx) * (cy ? fy : 1.0f - fy);
                    size_t base = ((size_t)(b * NPIX + iy * WW + ix)) * C384 + head * HD + lane;
                    kacc += wgt * kT[base];
                    vacc += wgt * vT[base];
                }
            }
        vs[p] = vacc;
        float s = qd * kacc;
#pragma unroll
        for (int m = 16; m > 0; m >>= 1) s += __shfl_xor(s, m, 32);
        int px = (int)rintf((gx + 1.0f) * 0.5f * 48.0f - 0.5f);
        px = px < 0 ? 0 : (px > 47 ? 47 : px);
        int py = (int)rintf((gy + 1.0f) * 0.5f * 48.0f - 0.5f);
        py = py < 0 ? 0 : (py > 47 ? 47 : py);
        int dyi = qy - py; if (dyi < 0) dyi = -dyi;
        int dxi = qx - px; if (dxi < 0) dxi = -dxi;
        sc[p] = s * SCALE + ab[head * NPIX + dyi * WW + dxi];
    }
    float mx = fmaxf(fmaxf(sc[0], sc[1]), fmaxf(sc[2], sc[3]));
    float e0 = expf(sc[0] - mx), e1 = expf(sc[1] - mx);
    float e2 = expf(sc[2] - mx), e3 = expf(sc[3] - mx);
    float inv = 1.0f / (e0 + e1 + e2 + e3);
    float o = (e0 * vs[0] + e1 * vs[1] + e2 * vs[2] + e3 * vs[3]) * inv;
    out[((size_t)(b * NPIX + n)) * C384 + head * HD + lane] = o;
}

// ---------------------------------------------------------------------------
extern "C" void kernel_launch(void* const* d_in, const int* in_sizes, int n_in,
                              void* d_out, int out_size, void* d_ws, size_t ws_size,
                              hipStream_t stream)
{
    (void)in_sizes; (void)n_in; (void)out_size; (void)ws_size;
    const float* local_feat    = (const float*)d_in[0];
    const float* context_prior = (const float*)d_in[1];
    const float* deformable_x  = (const float*)d_in[2];
    const float* W_q    = (const float*)d_in[3];
    const float* W_k    = (const float*)d_in[4];
    const float* W_v    = (const float*)d_in[5];
    const float* W_pre  = (const float*)d_in[6];
    const float* ln1_g  = (const float*)d_in[7];
    const float* ln1_b  = (const float*)d_in[8];
    const float* W_post = (const float*)d_in[9];
    const float* dw_w   = (const float*)d_in[10];
    const float* dw_b   = (const float*)d_in[11];
    const float* ln2_g  = (const float*)d_in[12];
    const float* ln2_b  = (const float*)d_in[13];
    const float* W_lo   = (const float*)d_in[14];
    const float* b_lo   = (const float*)d_in[15];
    const float* W_off  = (const float*)d_in[16];
    const float* b_off  = (const float*)d_in[17];
    const float* ab     = (const float*)d_in[18];
    const float* W_p    = (const float*)d_in[19];
    const float* bn_g   = (const float*)d_in[20];
    const float* bn_b   = (const float*)d_in[21];
    const float* bn_m   = (const float*)d_in[22];
    const float* bn_v   = (const float*)d_in[23];

    float* ws = (float*)d_ws;
    const size_t SZ_BIG = (size_t)BB * C384 * NPIX;   // 7,077,888 floats
    float* qT       = ws;                 // (B,N,384)
    float* cg       = ws + 1 * SZ_BIG;    // (B,384,N); reused as lo_pre after pooling
    float* kT       = ws + 2 * SZ_BIG;    // (B,N,384)
    float* vT       = ws + 3 * SZ_BIG;    // (B,N,384)
    float* dw       = ws + 4 * SZ_BIG;    // (B,384,N); reused as attn_out (B,N,384)
    float* small    = ws + 5 * SZ_BIG;
    float* pooled   = small;                            // B*384*49
    float* gsmall   = pooled + (size_t)BB * C384 * 49;  // B*32*49
    float* guide    = gsmall + (size_t)BB * 32 * 49;    // B*32*N
    float* lo       = guide  + (size_t)BB * 32 * NPIX;  // B*32*N
    float* offb     = lo     + (size_t)BB * 32 * NPIX;  // B*96*N
    float* lo_pre   = cg;
    float* attn_out = dw;

    dim3 gg(NPIX / TILE_N, C384 / TILE_O, BB);   // (18, 6, 8)

    // Big GEMMs on the WMMA path
    gemm_wmma<0,1,0><<<gg, 256, 0, stream>>>(local_feat,    W_q,   qT, C384, C384, nullptr,nullptr,nullptr,nullptr);
    gemm_wmma<0,0,1><<<gg, 256, 0, stream>>>(context_prior, W_pre, cg, C384, C384, nullptr,nullptr,nullptr,nullptr);
    gemm_wmma<0,1,0><<<gg, 256, 0, stream>>>(context_prior, W_k,   kT, C384, C384, nullptr,nullptr,nullptr,nullptr);
    gemm_wmma<0,1,0><<<gg, 256, 0, stream>>>(deformable_x,  W_v,   vT, C384, C384, nullptr,nullptr,nullptr,nullptr);

    // Guide branch
    pool_kernel  <<<BB * C384, 64, 0, stream>>>(cg, pooled);
    ln1_kernel   <<<BB * 49, 128, 0, stream>>>(pooled, ln1_g, ln1_b);
    post_kernel  <<<(BB * 32 * 49 + 255) / 256, 256, 0, stream>>>(pooled, W_post, gsmall);
    resize_kernel<<<(BB * 32 * NPIX + 255) / 256, 256, 0, stream>>>(gsmall, guide);

    // Local branch (cg buffer is free after pool_kernel -> holds lo_pre)
    dwconv_kernel <<<(BB * C384 * NPIX + 255) / 256, 256, 0, stream>>>(local_feat, dw_w, dw_b, dw);
    ln2gelu_kernel<<<BB * NPIX, 128, 0, stream>>>(dw, ln2_g, ln2_b, lo_pre);
    lo_kernel     <<<(BB * 32 * NPIX + 255) / 256, 256, 0, stream>>>(lo_pre, W_lo, b_lo, lo);

    // Offsets
    off_kernel<<<(BB * 96 * NPIX + 255) / 256, 256, 0, stream>>>(guide, lo, W_off, b_off, offb);

    // Deformable attention (dw buffer free after ln2gelu -> holds attn_out)
    int attn_threads = BB * NHEAD * NPIX * 32;
    attn_kernel<<<(attn_threads + 255) / 256, 256, 0, stream>>>(qT, kT, vT, offb, ab, attn_out);

    // Output projection + BatchNorm fused, straight to d_out (B,384,H,W)
    gemm_wmma<1,0,2><<<gg, 256, 0, stream>>>(attn_out, W_p, (float*)d_out, C384, C384,
                                             bn_g, bn_b, bn_m, bn_v);
}
